// Attention_20392504721471
// MI455X (gfx1250) — matrix operand order repaired and checked by reading, not compiled, not run on hardware
//
#include <hip/hip_runtime.h>
#include <hip/hip_bf16.h>
#include <math.h>

#define B_  16
#define F_  128
#define S_  256
#define H_  256   // == QS == CS

typedef __attribute__((ext_vector_type(16))) _Float16 v16h;
typedef __attribute__((ext_vector_type(8)))  float    v8f;

__device__ __forceinline__ float fast_tanh(float x) {
#if __has_builtin(__builtin_amdgcn_tanhf)
    return __builtin_amdgcn_tanhf(x);   // v_tanh_f32 on gfx1250
#else
    return tanhf(x);
#endif
}

template <int BASE>
__device__ __forceinline__ void cvt8(v16h& d, float4 lo, float4 hi) {
    d[BASE + 0] = (_Float16)lo.x; d[BASE + 1] = (_Float16)lo.y;
    d[BASE + 2] = (_Float16)lo.z; d[BASE + 3] = (_Float16)lo.w;
    d[BASE + 4] = (_Float16)hi.x; d[BASE + 5] = (_Float16)hi.y;
    d[BASE + 6] = (_Float16)hi.z; d[BASE + 7] = (_Float16)hi.w;
}

// P[r, n] = sum_k X[r, k] * W[n, k]       (X: R x 256, W: 256 x 256, P: R x 256)
// One wave32 per 16x16 output tile; K=256 swept in 8 steps of v_wmma_f32_16x16x32_f16.
__global__ __launch_bounds__(32)
void wmma_project(const float* __restrict__ X, const float* __restrict__ W,
                  float* __restrict__ P) {
    const int K = H_, N = H_;
    const int nTilesN  = N >> 4;                       // 16
    const int rowBase  = (blockIdx.x / nTilesN) << 4;
    const int colBase  = (blockIdx.x % nTilesN) << 4;
    const int lane     = threadIdx.x;                  // 0..31
    const int hi       = lane >> 4;                    // 0 or 1
    const int l15      = lane & 15;

    const float* arow = X + (size_t)(rowBase + l15) * K;   // A: row = M
    const float* brow = W + (size_t)(colBase + l15) * K;   // B: col n = W row n (over k)

    v8f c = {};
#pragma unroll
    for (int k0 = 0; k0 < K; k0 += 32) {
        // A-matrix 16x32 f16 layout (ISA 7.12.2):
        //   lanes 0-15 : halves 0-7 = K[k0..k0+7],   halves 8-15 = K[k0+16..k0+23]
        //   lanes 16-31: halves 0-7 = K[k0+8..k0+15], halves 8-15 = K[k0+24..k0+31]
        const int oa0 = k0 + (hi ? 8 : 0);
        const int oa1 = k0 + 16 + (hi ? 8 : 0);
        v16h a;
        cvt8<0>(a, *(const float4*)(arow + oa0), *(const float4*)(arow + oa0 + 4));
        cvt8<8>(a, *(const float4*)(arow + oa1), *(const float4*)(arow + oa1 + 4));

        // B-matrix 32x16 f16 layout (same K order as A):
        //   lane n (0-15): col n, halves 0-7 = K[k0..k0+7],   halves 8-15 = K[k0+16..k0+23]
        //   lane 16+n    : col n, halves 0-7 = K[k0+8..k0+15], halves 8-15 = K[k0+24..k0+31]
        const int ob = k0 + (hi ? 8 : 0);
        v16h b;
        cvt8<0>(b, *(const float4*)(brow + ob),      *(const float4*)(brow + ob + 4));
        cvt8<8>(b, *(const float4*)(brow + ob + 16), *(const float4*)(brow + ob + 20));

        c = __builtin_amdgcn_wmma_f32_16x16x32_f16(
                /*neg_a=*/false, a, /*neg_b=*/false, b,
                /*c_mod=*/(short)0, c, /*reuse_a=*/false, /*reuse_b=*/false);
    }

    // D layout: VGPR r -> M = r (lanes 0-15) / M = 8+r (lanes 16-31); N = lane&15
    float* prow = P + (size_t)(rowBase + (hi ? 8 : 0)) * N + colBase + l15;
#pragma unroll
    for (int r = 0; r < 8; ++r)
        prow[(size_t)r * N] = c[r];
}

// One 256-thread block per (b,f). Thread t owns s = t.
// energy[s] = sum_h v[h] * tanh(qp[b,f,h] + mp[b,s,h]); then softmax over s.
__global__ __launch_bounds__(256)
void energy_softmax(const float* __restrict__ qp, const float* __restrict__ mp,
                    const float* __restrict__ v, const unsigned char* __restrict__ mask,
                    float* __restrict__ out) {
    __shared__ float4 sq4[H_ / 4];
    __shared__ float4 sv4[H_ / 4];
    __shared__ float  sred[S_];

    const int bf = blockIdx.x;           // 0 .. B*F-1
    const int b  = bf / F_;
    const int s  = threadIdx.x;          // 0 .. 255

    ((float*)sq4)[s] = qp[(size_t)bf * H_ + s];
    ((float*)sv4)[s] = v[s];
    __syncthreads();

    const float4* mrow = (const float4*)(mp + (size_t)(b * S_ + s) * H_);
    float e = 0.0f;
#pragma unroll 8
    for (int h4 = 0; h4 < H_ / 4; ++h4) {
        float4 m  = mrow[h4];
        float4 q  = sq4[h4];
        float4 vv = sv4[h4];
        e = fmaf(vv.x, fast_tanh(q.x + m.x), e);
        e = fmaf(vv.y, fast_tanh(q.y + m.y), e);
        e = fmaf(vv.z, fast_tanh(q.z + m.z), e);
        e = fmaf(vv.w, fast_tanh(q.w + m.w), e);
    }

    if (mask[(size_t)bf * S_ + s])
        e = -INFINITY;

    // block softmax over s (256 threads = 8 wave32)
    sred[s] = e;
    __syncthreads();
#pragma unroll
    for (int off = 128; off > 0; off >>= 1) {
        if (s < off) sred[s] = fmaxf(sred[s], sred[s + off]);
        __syncthreads();
    }
    const float mx = sred[0];
    __syncthreads();

    const float ex = __expf(e - mx);     // v_exp_f32
    sred[s] = ex;
    __syncthreads();
#pragma unroll
    for (int off = 128; off > 0; off >>= 1) {
        if (s < off) sred[s] = sred[s] + sred[s + off];
        __syncthreads();
    }
    const float inv = 1.0f / sred[0];

    out[(size_t)bf * S_ + s] = ex * inv;
}

extern "C" void kernel_launch(void* const* d_in, const int* in_sizes, int n_in,
                              void* d_out, int out_size, void* d_ws, size_t ws_size,
                              hipStream_t stream) {
    (void)in_sizes; (void)n_in; (void)out_size; (void)ws_size;
    const float*         query  = (const float*)d_in[0];          // (B,F,QS)
    const float*         memory = (const float*)d_in[1];          // (B,S,CS)
    const float*         W_q    = (const float*)d_in[2];          // (H,QS)
    const float*         W_c    = (const float*)d_in[3];          // (H,CS)
    const float*         v      = (const float*)d_in[4];          // (H,)
    const unsigned char* mask   = (const unsigned char*)d_in[5];  // (B,F,S) bool
    float*               out    = (float*)d_out;                  // (B,F,S)

    float* qp = (float*)d_ws;                         // (B*F, H) = 2 MB
    float* mp = qp + (size_t)B_ * F_ * H_;            // (B*S, H) = 4 MB

    // qp = query @ W_q^T : 2048x256 output -> 128*16 tiles
    wmma_project<<<dim3((B_ * F_ / 16) * (H_ / 16)), dim3(32), 0, stream>>>(query, W_q, qp);
    // mp = memory @ W_c^T : 4096x256 output -> 256*16 tiles
    wmma_project<<<dim3((B_ * S_ / 16) * (H_ / 16)), dim3(32), 0, stream>>>(memory, W_c, mp);
    // energy + softmax
    energy_softmax<<<dim3(B_ * F_), dim3(S_), 0, stream>>>(qp, mp, v, mask, out);
}
